// NanoGpt_21208548508360
// MI455X (gfx1250) — matrix operand order repaired and checked by reading, not compiled, not run on hardware
//
#include <hip/hip_runtime.h>
#include <math.h>

typedef _Float16 h16;
typedef __attribute__((ext_vector_type(16))) _Float16 v16h;
typedef __attribute__((ext_vector_type(8)))  float    v8f;
typedef __attribute__((ext_vector_type(2)))  _Float16 h2;

#define B_SZ   2
#define S_SZ   1024
#define D_SZ   768
#define L_SZ   6
#define V_SZ   50257
#define M_ROWS (B_SZ * S_SZ)   // 2048

// ---------------- embedding: h = tok_emb[x] + pos_emb ----------------
__global__ void embed_kernel(const int* __restrict__ x, const float* __restrict__ tok,
                             const float* __restrict__ pos, float* __restrict__ h) {
  int idx = blockIdx.x * blockDim.x + threadIdx.x;
  if (idx >= M_ROWS * D_SZ) return;
  int row = idx / D_SZ, d = idx - row * D_SZ;
  int s = row % S_SZ;
  int t = x[row];
  h[idx] = tok[(long)t * D_SZ + d] + pos[(long)s * D_SZ + d];
}

// ---------------- layernorm: one block (256 thr) per row ----------------
__global__ void ln_kernel(const float* __restrict__ in, const float* __restrict__ g,
                          const float* __restrict__ b, float* __restrict__ out) {
  __shared__ float s1[256], s2[256];
  int row = blockIdx.x;
  const float* xr = in + (long)row * D_SZ;
  float sum = 0.f, sq = 0.f;
  for (int i = threadIdx.x; i < D_SZ; i += 256) { float v = xr[i]; sum += v; sq += v * v; }
  s1[threadIdx.x] = sum; s2[threadIdx.x] = sq;
  __syncthreads();
  for (int off = 128; off > 0; off >>= 1) {
    if (threadIdx.x < off) { s1[threadIdx.x] += s1[threadIdx.x + off]; s2[threadIdx.x] += s2[threadIdx.x + off]; }
    __syncthreads();
  }
  float mean = s1[0] * (1.0f / D_SZ);
  float var  = s2[0] * (1.0f / D_SZ) - mean * mean;
  float inv  = rsqrtf(var + 1e-5f);
  for (int i = threadIdx.x; i < D_SZ; i += 256)
    out[(long)row * D_SZ + i] = (xr[i] - mean) * inv * g[i] + b[i];
}

// ---------------- generic WMMA GEMM ----------------
// C[M,N] = act(A[M,K] @ B[K, bcol:bcol+N] + bias) + resid
// Block tile 64(M) x 128(N), 8 waves (2x4), each wave owns 32x32:
// 2 A-frags x 2 B-frags -> 4 v_wmma_f32_16x16x32_f16 per K-step per wave.
#define BM 64
#define BN 128
#define BK 32
#define LDA_S 34   // halves per As row (pad keeps pairs dword-aligned, frag reads conflict-free)
#define LDB_S 34   // halves per Bs row (Bs transposed: [n][k])

template<bool HAS_BIAS, bool GELU, bool HAS_RESID>
__global__ __launch_bounds__(256)
void wmma_gemm_kernel(const float* __restrict__ A, const float* __restrict__ B,
                      const float* __restrict__ bias, const float* __restrict__ resid,
                      float* __restrict__ C, int M, int N, int K, long ldb, int bcol)
{
  __shared__ h16 As[BM * LDA_S];   // [64][34]
  __shared__ h16 Bs[BN * LDB_S];   // [128][34], n-major / k-inner

  const int tid  = threadIdx.x;
  const int lane = tid & 31;
  const int wid  = tid >> 5;       // 0..7
  const int waveM = wid >> 2;      // 0..1  -> M offset waveM*32
  const int waveN = wid & 3;       // 0..3  -> N offset waveN*32
  const int half = lane >> 4;      // 0/1
  const int lm   = lane & 15;

  const int blockM = blockIdx.y * BM;
  const int blockN = blockIdx.x * BN;
  const bool fullN = (blockN + BN) <= N;   // uniform across block

  v8f acc00 = {}, acc01 = {}, acc10 = {}, acc11 = {};

  // ---- hoisted per-thread staging pointers (bumped each K-step, no muls in loop)
  const int ar = tid >> 3;          // 0..31 (A tile rows ar, ar+32)
  const int ac = (tid & 7) << 2;    // 0..28
  const float* aptr0 = A + (long)(blockM + ar) * K + ac;
  const float* aptr1 = aptr0 + (long)32 * K;

  const int bk  = tid >> 5;         // base k row within tile for this thread
  const int bl  = tid & 31;         // lane-contiguous column base
  const float* bptr0 = B + (long)bk * ldb + bcol + blockN + bl;        // rows bk + 8p
  const float* bptr1 = bptr0 + 8 * ldb;
  const float* bptr2 = bptr0 + 16 * ldb;
  const float* bptr3 = bptr0 + 24 * ldb;
  const long bstep = (long)BK * ldb;

  h16* const asdst = &As[ar * LDA_S + ac];
  h16* const bsdst = &Bs[bl * LDB_S + bk];  // column bl, +32*LDB_S per i

  for (int k0 = 0; k0 < K; k0 += BK) {
    // ---- stage A tile 64x32 (fp32 -> f16), coalesced b128 loads
    {
      float4 v0 = *(const float4*)aptr0;
      float4 v1 = *(const float4*)aptr1;
      asdst[0] = (h16)v0.x; asdst[1] = (h16)v0.y; asdst[2] = (h16)v0.z; asdst[3] = (h16)v0.w;
      h16* d1 = asdst + 32 * LDA_S;
      d1[0] = (h16)v1.x; d1[1] = (h16)v1.y; d1[2] = (h16)v1.z; d1[3] = (h16)v1.w;
    }
    if (k0 + BK < K)  // gfx1250 global_prefetch for next A tile
      __builtin_prefetch(aptr0 + BK, 0, 1);
    aptr0 += BK; aptr1 += BK;

    // ---- stage B tile 32x128 transposed into Bs[n][k]
    // every global load is a 128B lane-contiguous line; offsets 0/32/64/96 are immediates.
    if (fullN) {
      #pragma unroll
      for (int i = 0; i < 4; ++i) {
        bsdst[(32 * i) * LDB_S] = (h16)bptr0[32 * i];
        bsdst[(32 * i) * LDB_S + 8]  = (h16)bptr1[32 * i];
        bsdst[(32 * i) * LDB_S + 16] = (h16)bptr2[32 * i];
        bsdst[(32 * i) * LDB_S + 24] = (h16)bptr3[32 * i];
      }
    } else {
      // cold edge path (only the last N-block of the logits GEMM)
      #pragma unroll 1
      for (int i = 0; i < 4; ++i) {
        int ncol = blockN + bl + 32 * i;
        int coff = ((ncol < N) ? ncol : (N - 1)) - blockN - bl;  // clamped, unconditional loads
        float v0 = bptr0[coff], v1 = bptr1[coff], v2 = bptr2[coff], v3 = bptr3[coff];
        if (ncol >= N) { v0 = 0.f; v1 = 0.f; v2 = 0.f; v3 = 0.f; }
        bsdst[(32 * i) * LDB_S]      = (h16)v0;
        bsdst[(32 * i) * LDB_S + 8]  = (h16)v1;
        bsdst[(32 * i) * LDB_S + 16] = (h16)v2;
        bsdst[(32 * i) * LDB_S + 24] = (h16)v3;
      }
    }
    bptr0 += bstep; bptr1 += bstep; bptr2 += bstep; bptr3 += bstep;
    __syncthreads();

    // ---- build fragments per ISA wave32 layouts
    v16h a0, a1, b0, b1;
    #pragma unroll
    for (int i = 0; i < 8; ++i) {
      // A 16x32 f16: VGPR i holds K pair base (2i | 2i+8) + 8*half, row = lane&15
      int ka = ((i < 4) ? 2 * i : 2 * i + 8) + 8 * half;
      h2 pa0 = *(const h2*)&As[(waveM * 32 + lm) * LDA_S + ka];
      a0[2 * i] = pa0.x; a0[2 * i + 1] = pa0.y;
      h2 pa1 = *(const h2*)&As[(waveM * 32 + 16 + lm) * LDA_S + ka];
      a1[2 * i] = pa1.x; a1[2 * i + 1] = pa1.y;
      // B 32x16 f16: VGPR i holds K pair base (2i | 2i+8) + 8*half, column = lane&15
      int kb = ((i < 4) ? 2 * i : 2 * i + 8) + 8 * half;
      h2 pb0 = *(const h2*)&Bs[(waveN * 32 + lm) * LDB_S + kb];
      b0[2 * i] = pb0.x; b0[2 * i + 1] = pb0.y;
      h2 pb1 = *(const h2*)&Bs[(waveN * 32 + 16 + lm) * LDB_S + kb];
      b1[2 * i] = pb1.x; b1[2 * i + 1] = pb1.y;
    }

    acc00 = __builtin_amdgcn_wmma_f32_16x16x32_f16(false, a0, false, b0, (short)0, acc00, false, false);
    acc01 = __builtin_amdgcn_wmma_f32_16x16x32_f16(false, a0, false, b1, (short)0, acc01, false, false);
    acc10 = __builtin_amdgcn_wmma_f32_16x16x32_f16(false, a1, false, b0, (short)0, acc10, false, false);
    acc11 = __builtin_amdgcn_wmma_f32_16x16x32_f16(false, a1, false, b1, (short)0, acc11, false, false);
    __syncthreads();
  }

  // ---- epilogue: D layout M = j + 8*half, N = lane&15
  auto apply = [&](int m, int n, float v) -> float {
    if (HAS_BIAS) v += bias[n];
    if (GELU)     v = 0.5f * v * (1.0f + erff(v * 0.70710678118654752f));
    if (HAS_RESID) v += resid[(long)m * N + n];
    return v;
  };
  const int m0 = blockM + waveM * 32 + half * 8;
  const int n0 = blockN + waveN * 32 + lm;
  if (fullN) {
    #pragma unroll
    for (int j = 0; j < 8; ++j) {
      C[(long)(m0 + j) * N + n0]           = apply(m0 + j,      n0,      acc00[j]);
      C[(long)(m0 + j) * N + n0 + 16]      = apply(m0 + j,      n0 + 16, acc01[j]);
      C[(long)(m0 + 16 + j) * N + n0]      = apply(m0 + 16 + j, n0,      acc10[j]);
      C[(long)(m0 + 16 + j) * N + n0 + 16] = apply(m0 + 16 + j, n0 + 16, acc11[j]);
    }
  } else {
    #pragma unroll 1
    for (int j = 0; j < 8; ++j) {
      if (n0 < N) {
        C[(long)(m0 + j) * N + n0]      = apply(m0 + j,      n0, acc00[j]);
        C[(long)(m0 + 16 + j) * N + n0] = apply(m0 + 16 + j, n0, acc10[j]);
      }
      if (n0 + 16 < N) {
        C[(long)(m0 + j) * N + n0 + 16]      = apply(m0 + j,      n0 + 16, acc01[j]);
        C[(long)(m0 + 16 + j) * N + n0 + 16] = apply(m0 + 16 + j, n0 + 16, acc11[j]);
      }
    }
  }
}

// ---------------- orchestration ----------------
extern "C" void kernel_launch(void* const* d_in, const int* in_sizes, int n_in,
                              void* d_out, int out_size, void* d_ws, size_t ws_size,
                              hipStream_t stream) {
  const int*   x      = (const int*)  d_in[0];
  const float* tok    = (const float*)d_in[1];
  const float* pos    = (const float*)d_in[2];
  const float* ln1_g  = (const float*)d_in[3];
  const float* ln1_b  = (const float*)d_in[4];
  const float* attn_w = (const float*)d_in[5];
  const float* attn_b = (const float*)d_in[6];
  const float* proj_w = (const float*)d_in[7];
  const float* proj_b = (const float*)d_in[8];
  const float* ln2_g  = (const float*)d_in[9];
  const float* ln2_b  = (const float*)d_in[10];
  const float* mlp_w1 = (const float*)d_in[11];
  const float* mlp_b1 = (const float*)d_in[12];
  const float* mlp_w2 = (const float*)d_in[13];
  const float* mlp_b2 = (const float*)d_in[14];
  const float* lnf_g  = (const float*)d_in[15];
  const float* lnf_b  = (const float*)d_in[16];
  const float* out_w  = (const float*)d_in[17];

  float* h  = (float*)d_ws;
  float* t1 = h  + (size_t)M_ROWS * D_SZ;              // LN outputs
  float* t2 = t1 + (size_t)M_ROWS * D_SZ;              // V / GELU activations (2048 x 3072 max)

  const int D = D_SZ, F = 4 * D_SZ;

  embed_kernel<<<(M_ROWS * D_SZ + 255) / 256, 256, 0, stream>>>(x, tok, pos, h);

  dim3 blk(256);
  dim3 gD((D + BN - 1) / BN, M_ROWS / BM);
  dim3 gF((F + BN - 1) / BN, M_ROWS / BM);
  dim3 gV((V_SZ + BN - 1) / BN, M_ROWS / BM);

  for (int l = 0; l < L_SZ; ++l) {
    // a = LN1(h)
    ln_kernel<<<M_ROWS, 256, 0, stream>>>(h, ln1_g + (size_t)l * D, ln1_b + (size_t)l * D, t1);
    // v = a @ attn_w[l][:, 2D:3D] + attn_b[l][2D:]   (attention output == v; softmax rows sum to 1)
    wmma_gemm_kernel<true, false, false><<<gD, blk, 0, stream>>>(
        t1, attn_w + (size_t)l * D * 3 * D, attn_b + (size_t)l * 3 * D + 2 * D, nullptr,
        t2, M_ROWS, D, D, (long)3 * D, 2 * D);
    // h = v @ proj_w[l] + proj_b[l] + h
    wmma_gemm_kernel<true, false, true><<<gD, blk, 0, stream>>>(
        t2, proj_w + (size_t)l * D * D, proj_b + (size_t)l * D, h,
        h, M_ROWS, D, D, (long)D, 0);
    // m = LN2(h)
    ln_kernel<<<M_ROWS, 256, 0, stream>>>(h, ln2_g + (size_t)l * D, ln2_b + (size_t)l * D, t1);
    // g = gelu(m @ mlp_w1[l] + mlp_b1[l])
    wmma_gemm_kernel<true, true, false><<<gF, blk, 0, stream>>>(
        t1, mlp_w1 + (size_t)l * D * F, mlp_b1 + (size_t)l * F, nullptr,
        t2, M_ROWS, F, D, (long)F, 0);
    // h = g @ mlp_w2[l] + mlp_b2[l] + h
    wmma_gemm_kernel<true, false, true><<<gD, blk, 0, stream>>>(
        t2, mlp_w2 + (size_t)l * F * D, mlp_b2 + (size_t)l * D, h,
        h, M_ROWS, D, F, (long)D, 0);
  }

  // final LN + logits
  ln_kernel<<<M_ROWS, 256, 0, stream>>>(h, lnf_g, lnf_b, t1);
  wmma_gemm_kernel<false, false, false><<<gV, blk, 0, stream>>>(
      t1, out_w, nullptr, nullptr, (float*)d_out, M_ROWS, V_SZ, D, (long)V_SZ, 0);
}
